// GATModel_53180285059745
// MI455X (gfx1250) — matrix operand order repaired and checked by reading, not compiled, not run on hardware
//
#include <hip/hip_runtime.h>
#include <hip/hip_bf16.h>
#include <hip/hip_fp16.h>

// ---------------------------------------------------------------------------
// GAT model for MI455X (gfx1250): WMMA f16 GEMMs with 2x4 register tiling,
// TDM (tensor_load_to_lds) staging of the per-graph input row, scalar
// attention/softmax. One workgroup per graph (2048 blocks x 256 threads).
// ---------------------------------------------------------------------------

#define Bq    2048
#define Aq    8
#define OBSq  56
#define Nq    64        // A + OBS
#define Fq    6
#define Hq    8
#define Cq    64
#define HCq   512       // H*C
#define OUTq  2
#define NEFq  1
#define NFq   (Nq * Fq)                       // 384
#define FEATq (NFq + OBSq * 2 + OBSq * NEFq)  // 552
#define E1q   OBSq                            // 56
#define E2q   (OBSq + Nq)                     // 120
#define PAD   8
#define LDW   (HCq + PAD)                     // 520 halves per padded row

typedef __attribute__((ext_vector_type(16))) _Float16 v16h;
typedef __attribute__((ext_vector_type(8)))  _Float16 v8h;
typedef __attribute__((ext_vector_type(8)))  float    v8f;
typedef __attribute__((ext_vector_type(4)))  unsigned int u32x4;
typedef __attribute__((ext_vector_type(8)))  int      i32x8;
typedef __attribute__((ext_vector_type(4)))  int      i32x4;

// ---------------------------------------------------------------------------
// TDM: 1-D tensor copy of `nelem` f32 from global to LDS (issued per-wave,
// EXEC-independent). D# per CDNA5 ISA.
// ---------------------------------------------------------------------------
__device__ __forceinline__ void tdm_load_row_f32(const float* gsrc, void* ldsdst,
                                                 unsigned int nelem) {
  unsigned long long ga = (unsigned long long)gsrc;
  unsigned int lds = (unsigned int)(unsigned long long)ldsdst;  // low 32b = LDS offset
  u32x4 g0;
  g0[0] = 1u;                                          // count=1 (user descriptor)
  g0[1] = lds;                                         // lds_addr
  g0[2] = (unsigned int)ga;                            // global_addr[31:0]
  g0[3] = ((unsigned int)(ga >> 32) & 0x01FFFFFFu)     // global_addr[56:32]
          | (2u << 30);                                // type=2 ("image")
  i32x8 g1;
  g1[0] = (int)(2u << 16);                             // data_size=4B; wg_mask=0
  g1[1] = (int)((nelem & 0xFFFFu) << 16);              // tensor_dim0[15:0]
  g1[2] = (int)((nelem >> 16) | (1u << 16));           // tensor_dim0[31:16], tensor_dim1=1
  g1[3] = (int)((nelem & 0xFFFFu) << 16);              // tile_dim0 = nelem
  g1[4] = 0;                                           // tile_dim1/2 unused
  g1[5] = (int)nelem;                                  // tensor_dim0_stride[31:0]
  g1[6] = 0;
  g1[7] = 0;
  i32x4 z4 = 0;
#if __clang_major__ >= 23
  i32x8 z8 = 0;
  __builtin_amdgcn_tensor_load_to_lds(g0, g1, z4, z4, z8, 0);
#else
  __builtin_amdgcn_tensor_load_to_lds(g0, g1, z4, z4, 0);
#endif
}

// ---------------------------------------------------------------------------
// Kernel 0: transpose + f32->f16 convert W2 and W3 into WT (N-major):
//   WT[l][n][k] = W_l[k][n],  l in {0,1},  shapes 512x512.
// ---------------------------------------------------------------------------
__global__ void prep_wt_kernel(const float* __restrict__ W2,
                               const float* __restrict__ W3,
                               _Float16* __restrict__ WT) {
  int i = blockIdx.x * blockDim.x + threadIdx.x;   // 2*512*512 total
  if (i >= 2 * HCq * HCq) return;
  int l = i >> 18;
  int r = i & (HCq * HCq - 1);
  int n = r >> 9;
  int k = r & (HCq - 1);
  const float* W = l ? W3 : W2;
  WT[i] = (_Float16)W[k * HCq + n];
}

// ---------------------------------------------------------------------------
// WMMA tiled GEMM:  hX(64x512) = hA(64x512) @ W(512x512), WT f16 N-major.
// Wave -> (mhalf in 0..1, ng in 0..3): register tile of 2 M-tiles x 4 N-tiles
// (8 v8f accumulators), two passes of 4 N-tiles over ng's 8 N-tiles.
// Per K-step (32): 2 A frags (LDS) + 4 B frags (global, each element read
// twice per block, once per mhalf wave) feed 8 WMMAs.
// A layout (16-bit A 16x32): lane m row m; lanes 0-15 K {k0..+7, k0+16..+23},
// lanes 16-31 shifted +8.  B (32x16): lane n col n; lanes 0-15 K {k0..+7, k0+16..+23},
// lanes 16-31 shifted +8.  D: VGPR r -> M=r (+8 for lanes 16-31).
// ---------------------------------------------------------------------------
__device__ __forceinline__ void gemm_wmma_512(const _Float16* __restrict__ WT,
                                              _Float16 (*hA)[LDW],
                                              _Float16 (*hX)[LDW],
                                              int tid) {
  const int lane    = tid & 31;
  const int wave    = tid >> 5;
  const int mhalf   = wave & 1;          // 2 consecutive M-tiles
  const int ng      = wave >> 1;         // group of 8 N-tiles
  const int halfSel = (lane >> 4) & 1;
  const int l15     = lane & 15;

  for (int nc = 0; nc < 2; ++nc) {
    const int ntbase = ng * 8 + nc * 4;
    v8f acc[2][4];
#pragma unroll
    for (int m = 0; m < 2; ++m)
#pragma unroll
      for (int n = 0; n < 4; ++n) acc[m][n] = 0.0f;

#pragma unroll 2
    for (int k0 = 0; k0 < HCq; k0 += 32) {
      v16h aF[2];
#pragma unroll
      for (int m = 0; m < 2; ++m) {
        const int row = (mhalf * 2 + m) * 16 + l15;
        const int ka  = k0 + halfSel * 8;
        v8h lo = *(const v8h*)&hA[row][ka];
        v8h hi = *(const v8h*)&hA[row][ka + 16];
        aF[m] = __builtin_shufflevector(lo, hi,
            0, 1, 2, 3, 4, 5, 6, 7, 8, 9, 10, 11, 12, 13, 14, 15);
      }
      v16h bF[4];
#pragma unroll
      for (int n = 0; n < 4; ++n) {
        const int col = (ntbase + n) * 16 + l15;
        v8h blo = *(const v8h*)(WT + (size_t)col * HCq + k0 + halfSel * 8);
        v8h bhi = *(const v8h*)(WT + (size_t)col * HCq + k0 + halfSel * 8 + 16);
        bF[n] = __builtin_shufflevector(blo, bhi,
            0, 1, 2, 3, 4, 5, 6, 7, 8, 9, 10, 11, 12, 13, 14, 15);
      }
      if (k0 + 32 < HCq)
        __builtin_prefetch(WT + (size_t)(ntbase * 16 + l15) * HCq + k0 + 32, 0, 3);
#pragma unroll
      for (int m = 0; m < 2; ++m)
#pragma unroll
        for (int n = 0; n < 4; ++n)
          acc[m][n] = __builtin_amdgcn_wmma_f32_16x16x32_f16(
              false, aF[m], false, bF[n], (short)0, acc[m][n], false, false);
    }
    // write D tiles back to LDS as f16
#pragma unroll
    for (int m = 0; m < 2; ++m) {
      const int rowb = (mhalf * 2 + m) * 16 + halfSel * 8;
#pragma unroll
      for (int n = 0; n < 4; ++n) {
        const int colo = (ntbase + n) * 16 + l15;
#pragma unroll
        for (int r = 0; r < 8; ++r)
          hX[rowb + r][colo] = (_Float16)acc[m][n][r];
      }
    }
  }
}

// ---------------------------------------------------------------------------
// Main kernel: one block (256 threads) per graph.
// ---------------------------------------------------------------------------
__global__ __launch_bounds__(256) void gat_main_kernel(
    const float* __restrict__ tensor,
    const float* __restrict__ W1,
    const float* __restrict__ as1, const float* __restrict__ ad1,
    const float* __restrict__ ae1, const float* __restrict__ We1,
    const float* __restrict__ b1,
    const float* __restrict__ as2, const float* __restrict__ ad2,
    const float* __restrict__ ae2, const float* __restrict__ We2,
    const float* __restrict__ b2,
    const float* __restrict__ as3, const float* __restrict__ ad3,
    const float* __restrict__ ae3, const float* __restrict__ We3,
    const float* __restrict__ b3,
    const float* __restrict__ fc1_w, const float* __restrict__ fc1_b,
    const float* __restrict__ fc2_w, const float* __restrict__ fc2_b,
    const _Float16* __restrict__ WT,   // [2][512][512] f16 (N-major)
    float* __restrict__ out) {
  __shared__ _Float16 hA[Nq][LDW];     // layer input h (f16)       66560 B
  __shared__ _Float16 hX[Nq][LDW];     // xs = h @ W (f16)          66560 B
  __shared__ float gbuf[FEATq];        // TDM-staged input row       2208 B
  __shared__ float alS[Nq][Hq];
  __shared__ float alD[Nq][Hq];
  __shared__ float wgt[E2q][Hq];       // alpha -> softmax weights
  __shared__ float eav[E2q];
  __shared__ int   srcA[E2q];
  __shared__ int   dstA[E2q];
  __shared__ float dotWeAe[Hq];
  __shared__ float gemb[HCq];
  __shared__ float hfc[Aq][4 * Cq];

  const int tid = threadIdx.x;
  const int bb  = blockIdx.x;
  const float* __restrict__ g = tensor + (size_t)bb * Aq * FEATq;  // row 0

  // ---- TDM: stage the whole graph row (x | ei | ea) into LDS ----
  if (tid < 32) {                      // wave 0 issues the DMA (EXEC-independent)
    tdm_load_row_f32(g, gbuf, FEATq);
    __builtin_amdgcn_s_wait_tensorcnt(0);
  }
  __syncthreads();

  // ---- parse edges (masked to 0 when invalid, like the reference) ----
  for (int e = tid; e < OBSq; e += 256) {
    int s = (int)gbuf[NFq + e];
    int d = (int)gbuf[NFq + OBSq + e];
    bool valid = ((unsigned)s < Nq) && ((unsigned)d < Nq);
    srcA[e] = valid ? s : 0;
    dstA[e] = valid ? d : 0;
    eav[e]  = valid ? gbuf[NFq + 2 * OBSq + e] : 0.0f;
  }
  // ---- layer-1 transform xs1 = x @ W1 (K=6, scalar) -> hX ----
  for (int i = tid; i < Nq * HCq; i += 256) {
    int n = i >> 9, j = i & (HCq - 1);
    float acc = 0.0f;
#pragma unroll
    for (int f = 0; f < Fq; ++f) acc += gbuf[n * Fq + f] * W1[f * HCq + j];
    hX[n][j] = (_Float16)acc;
  }
  __syncthreads();

  // ---- self-loop edge attrs for layers 2/3 (entries 56..119) ----
  if (tid < Nq) {
    int n = tid;
    float cnt = 0.0f, s = 0.0f;
    for (int e = 0; e < OBSq; ++e)
      if (dstA[e] == n) { cnt += 1.0f; s += eav[e]; }
    eav[OBSq + n]  = s / fmaxf(cnt, 1.0f);
    srcA[OBSq + n] = n;
    dstA[OBSq + n] = n;
  }

  // ================= attention + aggregation (3 layers) =================
  const float* AS[3] = {as1, as2, as3};
  const float* AD[3] = {ad1, ad2, ad3};
  const float* AE[3] = {ae1, ae2, ae3};
  const float* WE[3] = {We1, We2, We3};
  const float* BI[3] = {b1, b2, b3};

  for (int layer = 0; layer < 3; ++layer) {
    const int E = (layer == 0) ? E1q : E2q;
    const float* a_s = AS[layer];
    const float* a_d = AD[layer];
    const float* a_e = AE[layer];
    const float* we  = WE[layer];
    const float* bia = BI[layer];

    // NEF==1: al_e[e,h] = ea[e] * dot(We_h, ae_h)
    if (tid < Hq) {
      float s = 0.0f;
      for (int c = 0; c < Cq; ++c)
        s += we[tid * Cq + c] * a_e[tid * Cq + c];
      dotWeAe[tid] = s;
    }
    // per-(node,head) source/dest logits from xs (hX)
    for (int i = tid; i < Nq * Hq; i += 256) {
      int n = i >> 3, h = i & 7;
      float s1 = 0.0f, s2 = 0.0f;
      for (int c = 0; c < Cq; ++c) {
        float v = (float)hX[n][h * Cq + c];
        s1 += v * a_s[h * Cq + c];
        s2 += v * a_d[h * Cq + c];
      }
      alS[n][h] = s1;
      alD[n][h] = s2;
    }
    __syncthreads();

    // segment softmax, one owner thread per (dst node, head) -> deterministic
    for (int i = tid; i < Nq * Hq; i += 256) {
      int n = i >> 3, h = i & 7;
      float m = -INFINITY;
      for (int e = 0; e < E; ++e)
        if (dstA[e] == n) {
          float al = alS[srcA[e]][h] + alD[n][h] + eav[e] * dotWeAe[h];
          al = (al > 0.0f) ? al : 0.2f * al;     // leaky_relu(.,0.2)
          wgt[e][h] = al;
          m = fmaxf(m, al);
        }
      if (!(m > -3.0e38f)) m = 0.0f;             // isfinite guard
      float den = 0.0f;
      for (int e = 0; e < E; ++e)
        if (dstA[e] == n) {
          float ex = __expf(wgt[e][h] - m);
          wgt[e][h] = ex;
          den += ex;
        }
      float inv = 1.0f / (den + 1e-16f);
      for (int e = 0; e < E; ++e)
        if (dstA[e] == n) wgt[e][h] *= inv;
    }
    __syncthreads();

    // aggregate: hA[n][:] = relu( sum_e w[e,h]*xs[src[e],:] + b )
    {
      const int n  = tid >> 2;
      const int j0 = (tid & 3) * 128;
      for (int c16 = 0; c16 < 8; ++c16) {
        const int jb = j0 + c16 * 16;
        const int h  = jb >> 6;
        float acc[16];
#pragma unroll
        for (int u = 0; u < 16; ++u) acc[u] = bia[jb + u];
        for (int e = 0; e < E; ++e) {
          if (dstA[e] == n) {
            float wv = wgt[e][h];
            int   sv = srcA[e];
            v8h x0 = *(const v8h*)&hX[sv][jb];
            v8h x1 = *(const v8h*)&hX[sv][jb + 8];
#pragma unroll
            for (int u = 0; u < 8; ++u) acc[u]     += wv * (float)x0[u];
#pragma unroll
            for (int u = 0; u < 8; ++u) acc[u + 8] += wv * (float)x1[u];
          }
        }
#pragma unroll
        for (int u = 0; u < 16; ++u)
          hA[n][jb + u] = (_Float16)fmaxf(acc[u], 0.0f);
      }
    }
    __syncthreads();

    // next layer's transform: xs = h @ W via WMMA (layers 2 and 3)
    if (layer < 2) {
      gemm_wmma_512(WT + (size_t)layer * HCq * HCq, hA, hX, tid);
      __syncthreads();
    }
  }

  // ================= readout head =================
  for (int c = tid; c < HCq; c += 256) {
    float s = 0.0f;
    for (int n = 0; n < Nq; ++n) s += (float)hA[n][c];
    gemb[c] = s * (1.0f / Nq);
  }
  __syncthreads();

  // fc1: comb = [agent(8x512) | gemb(512)] @ fc1_w(1024x256) + b, relu
  for (int idx = tid; idx < Aq * 4 * Cq; idx += 256) {
    int a = idx >> 8, j = idx & (4 * Cq - 1);
    float s = fc1_b[j];
    for (int i = 0; i < HCq; ++i)
      s += (float)hA[a][i] * fc1_w[i * (4 * Cq) + j];
    for (int i = 0; i < HCq; ++i)
      s += gemb[i] * fc1_w[(HCq + i) * (4 * Cq) + j];
    hfc[a][j] = fmaxf(s, 0.0f);
  }
  __syncthreads();

  // fc2: (8x256) @ (256x2) + b
  if (tid < Aq * OUTq) {
    int a = tid >> 1, o = tid & 1;
    float s = fc2_b[o];
    for (int j = 0; j < 4 * Cq; ++j)
      s += hfc[a][j] * fc2_w[j * OUTq + o];
    out[((size_t)bb * Aq + a) * OUTq + o] = s;
  }
}

// ---------------------------------------------------------------------------
extern "C" void kernel_launch(void* const* d_in, const int* in_sizes, int n_in,
                              void* d_out, int out_size, void* d_ws, size_t ws_size,
                              hipStream_t stream) {
  (void)in_sizes; (void)n_in; (void)out_size; (void)ws_size;
  const float* tensor = (const float*)d_in[0];
  const float* W1  = (const float*)d_in[1];
  const float* as1 = (const float*)d_in[2];
  const float* ad1 = (const float*)d_in[3];
  const float* ae1 = (const float*)d_in[4];
  const float* We1 = (const float*)d_in[5];
  const float* b1  = (const float*)d_in[6];
  const float* W2  = (const float*)d_in[7];
  const float* as2 = (const float*)d_in[8];
  const float* ad2 = (const float*)d_in[9];
  const float* ae2 = (const float*)d_in[10];
  const float* We2 = (const float*)d_in[11];
  const float* b2  = (const float*)d_in[12];
  const float* W3  = (const float*)d_in[13];
  const float* as3 = (const float*)d_in[14];
  const float* ad3 = (const float*)d_in[15];
  const float* ae3 = (const float*)d_in[16];
  const float* We3 = (const float*)d_in[17];
  const float* b3  = (const float*)d_in[18];
  const float* fc1_w = (const float*)d_in[19];
  const float* fc1_b = (const float*)d_in[20];
  const float* fc2_w = (const float*)d_in[21];
  const float* fc2_b = (const float*)d_in[22];

  _Float16* WT = (_Float16*)d_ws;   // 2*512*512*2 = 1 MiB of scratch

  prep_wt_kernel<<<(2 * HCq * HCq + 255) / 256, 256, 0, stream>>>(W2, W3, WT);

  gat_main_kernel<<<Bq, 256, 0, stream>>>(
      tensor, W1,
      as1, ad1, ae1, We1, b1,
      as2, ad2, ae2, We2, b2,
      as3, ad3, ae3, We3, b3,
      fc1_w, fc1_b, fc2_w, fc2_b,
      WT, (float*)d_out);
}
